// RBFNetwork_16527034155755
// MI455X (gfx1250) — matrix operand order repaired and checked by reading, not compiled, not run on hardware
//
#include <hip/hip_runtime.h>

#define B_SIZE 8192
#define N_SIZE 4096
#define D_SIZE 784
#define DPAD   800      // D padded to multiple of 32 (25 K-steps, no tail)
#define C_SIZE 10
#define NCHUNKS 8       // N / 512

// LDS arena layout (bytes): double-buffered staging, overlaid by res tile.
//   buf b at b*40960:  Xhi[128][40] | Xlo | Mhi | Mlo   (10240 B each)
//   res tile: float[128][132] = 67584 B  (overlays staging after K-loop)
#define TILE_B   10240
#define BUF_B    40960
#define RAW_B    81920

typedef __attribute__((ext_vector_type(16))) __bf16 bf16x16;
typedef __attribute__((ext_vector_type(8)))  __bf16 bf16x8;
typedef __attribute__((ext_vector_type(8)))  float  f32x8;
typedef __attribute__((ext_vector_type(4)))  float  f32x4;

__device__ __forceinline__ f32x8 zero8() {
    f32x8 z = {0.f, 0.f, 0.f, 0.f, 0.f, 0.f, 0.f, 0.f};
    return z;
}

__device__ __forceinline__ bf16x16 cat16(bf16x8 a, bf16x8 b) {
    return __builtin_shufflevector(a, b, 0, 1, 2, 3, 4, 5, 6, 7,
                                         8, 9, 10, 11, 12, 13, 14, 15);
}

// ---------------------------------------------------------------------------
// Kernel 1: one-time f32 -> bf16 hi/lo split into padded [rows][DPAD] arrays.
// One thread per 8 padded columns; pad columns (k >= 784) are zero.
// ---------------------------------------------------------------------------
__global__ void rbf_convert(const float* __restrict__ x, const float* __restrict__ mus,
                            __bf16* __restrict__ xhi, __bf16* __restrict__ xlo,
                            __bf16* __restrict__ mhi, __bf16* __restrict__ mlo) {
    const int tid = blockIdx.x * 256 + threadIdx.x;   // (B+N) * DPAD/8 threads
    const int row = tid / (DPAD / 8);
    const int kq  = (tid % (DPAD / 8)) * 8;
    const float* src;
    __bf16 *dhi, *dlo;
    if (row < B_SIZE) {
        src = x + (size_t)row * D_SIZE;
        dhi = xhi + (size_t)row * DPAD;
        dlo = xlo + (size_t)row * DPAD;
    } else {
        const int r = row - B_SIZE;
        src = mus + (size_t)r * D_SIZE;
        dhi = mhi + (size_t)r * DPAD;
        dlo = mlo + (size_t)r * DPAD;
    }
    bf16x8 hv, lv;
#pragma unroll
    for (int e = 0; e < 8; ++e) {
        const int k = kq + e;
        const float v = (k < D_SIZE) ? src[k] : 0.f;
        const __bf16 h = (__bf16)v;
        hv[e] = h;
        lv[e] = (__bf16)(v - (float)h);
    }
    *(bf16x8*)&dhi[kq] = hv;
    *(bf16x8*)&dlo[kq] = lv;
}

// ---------------------------------------------------------------------------
// Kernel 2: row norms ||x_i||^2 and ||mu_j||^2. One wave32 per row.
// ---------------------------------------------------------------------------
__global__ void rbf_norms(const float* __restrict__ x, const float* __restrict__ mus,
                          float* __restrict__ xn, float* __restrict__ mn) {
    const int wave = threadIdx.x >> 5;
    const int lane = threadIdx.x & 31;
    const int row  = blockIdx.x * 8 + wave;
    const float* src;
    float* dst;
    if (row < B_SIZE) {
        src = x + (size_t)row * D_SIZE;
        dst = xn + row;
    } else {
        const int r = row - B_SIZE;
        src = mus + (size_t)r * D_SIZE;
        dst = mn + r;
    }
    float s = 0.f;
    for (int k = lane; k < D_SIZE; k += 32) {
        const float v = src[k];
        s = fmaf(v, v, s);
    }
#pragma unroll
    for (int off = 16; off > 0; off >>= 1) s += __shfl_xor(s, off, 32);
    if (lane == 0) *dst = s;
}

// ---------------------------------------------------------------------------
// Kernel 3: fused RBF + second GEMM.
// 256 threads = 8 waves (2 M x 4 N). Block tile 128x128, 4 N-tiles per block.
// Staging: GLOBAL_LOAD_ASYNC_TO_LDS_B128 (ASYNCcnt), double-buffered.
// GEMM: split-bf16 hi/lo via v_wmma_f32_16x16x32_bf16, f32 accumulate.
// ---------------------------------------------------------------------------
__launch_bounds__(256)
__global__ void rbf_main(const __bf16* __restrict__ xhi, const __bf16* __restrict__ xlo,
                         const __bf16* __restrict__ mhi, const __bf16* __restrict__ mlo,
                         const float* __restrict__ sigs,
                         const float* __restrict__ W,
                         const float* __restrict__ xn,
                         const float* __restrict__ mn,
                         float* __restrict__ partial) {
    __shared__ __align__(16) char raw[RAW_B];

    const int tid   = threadIdx.x;
    const int wave  = tid >> 5;
    const int lane  = tid & 31;
    const int half  = lane >> 4;
    const int l16   = lane & 15;
    const int waveM = wave >> 2;   // 0..1 (64 rows each)
    const int waveN = wave & 3;    // 0..3 (32 cols each)

    const int row0   = blockIdx.x * 128;
    const int nchunk = blockIdx.y;

    const unsigned raw_lds = (unsigned)(uintptr_t)(void*)raw;  // LDS byte offset

    float outAcc[5];
#pragma unroll
    for (int ii = 0; ii < 5; ++ii) outAcc[ii] = 0.f;

    for (int nt = 0; nt < 4; ++nt) {
        const int col0 = nchunk * 512 + nt * 128;

        // --- per-thread async-copy addressing (8 b128 copies per wave/step)
        // chunk idx = t*256 + wave*32 + lane ; tile = t>>1 (Xhi,Xlo,Mhi,Mlo)
        unsigned lds_t[8];
        uint64_t ga_t[8];
#pragma unroll
        for (int t = 0; t < 8; ++t) {
            const int c    = ((t & 1) << 8) + wave * 32 + lane;  // 0..511
            const int rowl = c >> 2;                             // 0..127
            const int e    = c & 3;                              // 16B quarter
            const int tile = t >> 1;
            lds_t[t] = raw_lds + (unsigned)(tile * TILE_B + rowl * 80 + e * 16);
            const __bf16* base;
            size_t grow;
            if (tile == 0)      { base = xhi; grow = (size_t)(row0 + rowl); }
            else if (tile == 1) { base = xlo; grow = (size_t)(row0 + rowl); }
            else if (tile == 2) { base = mhi; grow = (size_t)(col0 + rowl); }
            else                { base = mlo; grow = (size_t)(col0 + rowl); }
            ga_t[t] = (uint64_t)(uintptr_t)(base + grow * DPAD + e * 8);
        }

        f32x8 acc[4][2];
#pragma unroll
        for (int i = 0; i < 4; ++i)
#pragma unroll
            for (int j = 0; j < 2; ++j) acc[i][j] = zero8();

        // prologue: stage K-step 0 into buffer 0
#pragma unroll
        for (int t = 0; t < 8; ++t) {
            const unsigned lo = lds_t[t];
            const uint64_t ga = ga_t[t];
            asm volatile("global_load_async_to_lds_b128 %0, %1, off"
                         :: "v"(lo), "v"(ga) : "memory");
        }

        for (int ks = 0; ks < 25; ++ks) {
            const int rb = ks & 1;
            if (ks < 24) {
                const int k0n = (ks + 1) * 32;
                const unsigned bofs = (unsigned)((rb ^ 1) * BUF_B);
#pragma unroll
                for (int t = 0; t < 8; ++t) {
                    const unsigned lo = lds_t[t] + bofs;
                    const uint64_t ga = ga_t[t] + (uint64_t)(k0n * 2);
                    asm volatile("global_load_async_to_lds_b128 %0, %1, off"
                                 :: "v"(lo), "v"(ga) : "memory");
                }
                // oldest 8 (current read buffer) complete when <= 8 outstanding
                asm volatile("s_wait_asynccnt 0x8" ::: "memory");
            } else {
                asm volatile("s_wait_asynccnt 0x0" ::: "memory");
            }
            __syncthreads();   // all waves' copies for buffer rb have landed

            const __bf16* pXhi = (const __bf16*)(raw + rb * BUF_B);
            const __bf16* pXlo = pXhi + TILE_B / 2;
            const __bf16* pMhi = pXhi + TILE_B;
            const __bf16* pMlo = pXhi + 3 * TILE_B / 2;

            // B fragments: lane l16 = N, K = 16*half + e (contiguous 16 bf16)
            bf16x16 bhi[2], blo[2];
#pragma unroll
            for (int j = 0; j < 2; ++j) {
                const int col = waveN * 32 + j * 16 + l16;
                const __bf16* ph = pMhi + col * 40 + 16 * half;
                const __bf16* pl = pMlo + col * 40 + 16 * half;
                bhi[j] = cat16(*(const bf16x8*)ph, *(const bf16x8*)(ph + 8));
                blo[j] = cat16(*(const bf16x8*)pl, *(const bf16x8*)(pl + 8));
            }
#pragma unroll
            for (int i = 0; i < 4; ++i) {
                // A fragment: lane l16 = M, K = 16*half + e (contiguous 16 bf16)
                const int row = waveM * 64 + i * 16 + l16;
                const __bf16* ph = pXhi + row * 40 + 16 * half;
                const __bf16* pl = pXlo + row * 40 + 16 * half;
                bf16x16 ahi = cat16(*(const bf16x8*)ph, *(const bf16x8*)(ph + 8));
                bf16x16 alo = cat16(*(const bf16x8*)pl, *(const bf16x8*)(pl + 8));
#pragma unroll
                for (int j = 0; j < 2; ++j) {
                    acc[i][j] = __builtin_amdgcn_wmma_f32_16x16x32_bf16(
                        false, ahi, false, bhi[j], (short)0, acc[i][j], false, false);
                    acc[i][j] = __builtin_amdgcn_wmma_f32_16x16x32_bf16(
                        false, ahi, false, blo[j], (short)0, acc[i][j], false, false);
                    acc[i][j] = __builtin_amdgcn_wmma_f32_16x16x32_bf16(
                        false, alo, false, bhi[j], (short)0, acc[i][j], false, false);
                }
            }
            __syncthreads();   // reads of buffer rb done before it is restaged
        } // ks

        // ---- epilogue: sq -> exp -> res tile (overlays staging LDS) ----
        float* resp = (float*)raw;   // [128][132]
        float coef[2], mnv[2];
#pragma unroll
        for (int j = 0; j < 2; ++j) {
            const int n = col0 + waveN * 32 + j * 16 + l16;
            const float s = sigs[n] * 5.0f;
            coef[j] = -0.5f / (s * s);
            mnv[j]  = mn[n];
        }
#pragma unroll
        for (int i = 0; i < 4; ++i) {
            const int rbase = waveM * 64 + i * 16 + 8 * half;
#pragma unroll
            for (int r = 0; r < 8; ++r) {
                const float xr = xn[row0 + rbase + r];
#pragma unroll
                for (int j = 0; j < 2; ++j) {
                    float sq = xr + mnv[j] - 2.0f * acc[i][j][r];
                    sq = fmaxf(sq, 0.f);
                    float res = __expf(sq * coef[j]);
                    if (!(res == res)) res = 0.f;   // NaN guard (matches reference)
                    resp[(rbase + r) * 132 + waveN * 32 + j * 16 + l16] = res;
                }
            }
        }
        __syncthreads();

        // ---- fused second GEMM: outAcc[m][c] += res_tile @ W^T slice ----
#pragma unroll
        for (int ii = 0; ii < 5; ++ii) {
            const int idx = ii * 256 + tid;   // 1280 = 128 rows x 10 cols
            const int m = idx / 10;
            const int c = idx - m * 10;
            const float* wrow = W + (size_t)c * N_SIZE + col0;
            float s = 0.f;
            for (int nl = 0; nl < 128; nl += 4) {
                f32x4 rv = *(const f32x4*)&resp[m * 132 + nl];
                f32x4 wv = *(const f32x4*)&wrow[nl];
                s = fmaf(rv[0], wv[0], s);
                s = fmaf(rv[1], wv[1], s);
                s = fmaf(rv[2], wv[2], s);
                s = fmaf(rv[3], wv[3], s);
            }
            outAcc[ii] += s;
        }
        __syncthreads();   // res tile consumed before next nt restages LDS
    } // nt

#pragma unroll
    for (int ii = 0; ii < 5; ++ii) {
        const int idx = ii * 256 + tid;
        const int m = idx / 10;
        const int c = idx - m * 10;
        partial[(size_t)nchunk * (B_SIZE * C_SIZE) +
                (size_t)(row0 + m) * C_SIZE + c] = outAcc[ii];
    }
}

// ---------------------------------------------------------------------------
// Kernel 4: reduce N-chunk partials + bias
// ---------------------------------------------------------------------------
__global__ void rbf_finalize(const float* __restrict__ partial,
                             const float* __restrict__ b,
                             float* __restrict__ out) {
    const int t = blockIdx.x * 256 + threadIdx.x;
    if (t >= B_SIZE * C_SIZE) return;
    float s = b[t % C_SIZE];
#pragma unroll
    for (int ch = 0; ch < NCHUNKS; ++ch)
        s += partial[(size_t)ch * (B_SIZE * C_SIZE) + t];
    out[t] = s;
}

extern "C" void kernel_launch(void* const* d_in, const int* in_sizes, int n_in,
                              void* d_out, int out_size, void* d_ws, size_t ws_size,
                              hipStream_t stream) {
    (void)in_sizes; (void)n_in; (void)out_size; (void)ws_size;
    const float* x    = (const float*)d_in[0];
    const float* mus  = (const float*)d_in[1];
    const float* sigs = (const float*)d_in[2];
    const float* W    = (const float*)d_in[3];
    const float* b    = (const float*)d_in[4];
    float* out = (float*)d_out;

    float* xn      = (float*)d_ws;                       // B floats
    float* mn      = xn + B_SIZE;                        // N floats
    float* partial = mn + N_SIZE;                        // NCHUNKS*B*C floats
    __bf16* xhi = (__bf16*)(partial + (size_t)NCHUNKS * B_SIZE * C_SIZE);
    __bf16* xlo = xhi + (size_t)B_SIZE * DPAD;
    __bf16* mhi = xlo + (size_t)B_SIZE * DPAD;
    __bf16* mlo = mhi + (size_t)N_SIZE * DPAD;           // total ws ~42 MB

    rbf_convert<<<(B_SIZE + N_SIZE) * (DPAD / 8) / 256, 256, 0, stream>>>(
        x, mus, xhi, xlo, mhi, mlo);

    rbf_norms<<<(B_SIZE + N_SIZE) / 8, 256, 0, stream>>>(x, mus, xn, mn);

    dim3 grid(B_SIZE / 128, NCHUNKS);   // 64 x 8 = 512 blocks
    rbf_main<<<grid, 256, 0, stream>>>(xhi, xlo, mhi, mlo, sigs, W, xn, mn, partial);

    rbf_finalize<<<(B_SIZE * C_SIZE + 255) / 256, 256, 0, stream>>>(partial, b, out);
}
